// WindowAttention_69664369541189
// MI455X (gfx1250) — matrix operand order repaired and checked by reading, not compiled, not run on hardware
//
#include <hip/hip_runtime.h>
#include <hip/hip_bf16.h>

typedef __attribute__((ext_vector_type(16))) _Float16 v16h;
typedef __attribute__((ext_vector_type(8)))  _Float16 v8h;
typedef __attribute__((ext_vector_type(8)))  float    v8f;

#define WS_   7
#define TOK   49
#define TPAD  64
#define CDIM  96
#define NH    3
#define HD    32
#define SCALE 0.17677669529663687f  /* 32^-0.5 */

// ---------------- shared memory layout (bytes) ----------------
#define XB_OFF    0        /* x   f16 [64][96]  */
#define QB_OFF    12288    /* q   f16 [64][96]  (later aliased by conv output) */
#define KB_OFF    24576    /* k   f16 [64][96]  */
#define VB_OFF    36864    /* v   f16 [64][96]  */
#define VTB_OFF   49152    /* v^T f16 [96][64]  */
#define QSTB_OFF  61440    /* qs^T f16 [96][64] */
#define KSTB_OFF  73728    /* ks^T f16 [96][64] */
#define SPB_OFF   86016    /* spatial attn out f16 [64][96] */
#define XCB_OFF   98304    /* spectral out f16 [64][96] */
#define ATT32_OFF 110592   /* attn scores f32 [64][64] | cattn f32 [3][32][32] | conv mid f16 */
#define ATT16_OFF 126976   /* attn probs f16 [64][64] | cattn f16 [3][32][32] */
#define C1_OFF    135168   /* conv1 f32 [9][96] */
#define C2_OFF    138624   /* conv2 f32 [9][96] */
#define BQ_OFF    142080
#define BK_OFF    142464
#define BV_OFF    142848
#define BF_OFF    143232
#define SMEM_BYTES 143616  /* 2 blocks / WGP at 320KB */

// ---------------- workspace layout (bytes) ----------------
#define WS_WQT   0
#define WS_WKT   18432
#define WS_WVT   36864
#define WS_WQSPT 55296
#define WS_WKSPT 73728
#define WS_WF1T  92160
#define WS_WF2T  110592
#define WS_WBOTT 129024
#define WS_BIASF 147456
#define WS_RPB   147840    /* [3][64][64] f32, end = 196992 */

// A fragment (16x32 f16, M x K) from row-major [rows][ld] halves.
// Lane l: M = l&15; halves 0..7 -> K = kbase..kbase+7, halves 8..15 -> K = kbase+16..kbase+23,
// kbase = k0 + (l<16 ? 0 : 8)   (CDNA5 ISA 7.12.2, 16-bit A 16x32 layout)
__device__ __forceinline__ v16h a_frag(const _Float16* base, int row0, int ld, int k0) {
  const int lane = threadIdx.x & 31;
  const int m  = lane & 15;
  const int kb = k0 + ((lane >> 4) << 3);
  const _Float16* p = base + (row0 + m) * ld + kb;
  v8h lo = *(const v8h*)(p);
  v8h hi = *(const v8h*)(p + 16);
  v16h r;
#pragma unroll
  for (int i = 0; i < 8; ++i) { r[i] = lo[i]; r[i + 8] = hi[i]; }
  return r;
}

// B fragment (32x16 f16, K x N) from [ncol][ld] layout (K contiguous per column).
// Lane l: N = l&15; halves 0..7 -> K = kbase..kbase+7, halves 8..15 -> K = kbase+16..kbase+23, kbase = k0 + (l<16 ? 0 : 8).
__device__ __forceinline__ v16h b_frag(const _Float16* base, int n0, int ld, int k0) {
  const int lane = threadIdx.x & 31;
  const int n  = lane & 15;
  const int kb = k0 + ((lane >> 4) << 3);
  const _Float16* p = base + (n0 + n) * ld + kb;
  v8h lo = *(const v8h*)(p);
  v8h hi = *(const v8h*)(p + 16);
  v16h r;
#pragma unroll
  for (int i = 0; i < 8; ++i) { r[i] = lo[i]; r[i + 8] = hi[i]; }
  return r;
}

#define WMMA(A,B,C) __builtin_amdgcn_wmma_f32_16x16x32_f16(false,(A),false,(B),(short)0,(C),false,false)

// ================= prep kernel: weight transpose/convert/fuse =================
__global__ __launch_bounds__(256) void wattn_prep(
    const float* __restrict__ rpb_table,
    const float* __restrict__ wq, const float* __restrict__ wk, const float* __restrict__ wv,
    const float* __restrict__ w_ps, const float* __restrict__ b_ps,
    const float* __restrict__ wq_sp, const float* __restrict__ wk_sp,
    const float* __restrict__ w_pc, const float* __restrict__ b_pc,
    const float* __restrict__ w_proj, const float* __restrict__ b_proj,
    char* __restrict__ ws)
{
  _Float16* wqT   = (_Float16*)(ws + WS_WQT);
  _Float16* wkT   = (_Float16*)(ws + WS_WKT);
  _Float16* wvT   = (_Float16*)(ws + WS_WVT);
  _Float16* wqspT = (_Float16*)(ws + WS_WQSPT);
  _Float16* wkspT = (_Float16*)(ws + WS_WKSPT);
  _Float16* wf1T  = (_Float16*)(ws + WS_WF1T);
  _Float16* wf2T  = (_Float16*)(ws + WS_WF2T);
  _Float16* wbotT = (_Float16*)(ws + WS_WBOTT);
  float*    biasf = (float*)   (ws + WS_BIASF);
  float*    rpbf  = (float*)   (ws + WS_RPB);

  const int gid = blockIdx.x * blockDim.x + threadIdx.x;
  const int gs  = gridDim.x * blockDim.x;

  // plain transposed f16 copies: wT[n][k] = W[k][n]
  for (int i = gid; i < 5 * CDIM * CDIM; i += gs) {
    int mi = i / (CDIM * CDIM), r = i % (CDIM * CDIM);
    int n = r / CDIM, k = r % CDIM;
    const float* W = (mi==0)?wq:(mi==1)?wk:(mi==2)?wv:(mi==3)?wq_sp:wk_sp;
    _Float16* D    = (mi==0)?wqT:(mi==1)?wkT:(mi==2)?wvT:(mi==3)?wqspT:wkspT;
    D[n*CDIM + k] = (_Float16)W[k*CDIM + n];
  }
  // fused weights: Wf1 = w_ps @ Wproj[0:96], Wf2 = w_pc @ Wproj[96:192]; WbotT = Wproj[96:192]^T
  for (int i = gid; i < 3 * CDIM * CDIM; i += gs) {
    int mi = i / (CDIM * CDIM), r = i % (CDIM * CDIM);
    int n = r / CDIM, k = r % CDIM;
    if (mi == 2) {
      wbotT[n*CDIM + k] = (_Float16)w_proj[(CDIM + k)*CDIM + n];
    } else {
      const float* A = (mi==0) ? w_ps : w_pc;
      int off = (mi==0) ? 0 : CDIM;
      float s = 0.0f;
      for (int j = 0; j < CDIM; ++j) s += A[k*CDIM + j] * w_proj[(off + j)*CDIM + n];
      ((mi==0)?wf1T:wf2T)[n*CDIM + k] = (_Float16)s;
    }
  }
  // fused bias
  for (int c = gid; c < CDIM; c += gs) {
    float s = b_proj[c];
    for (int j = 0; j < CDIM; ++j)
      s += b_ps[j]*w_proj[j*CDIM + c] + b_pc[j]*w_proj[(CDIM + j)*CDIM + c];
    biasf[c] = s;
  }
  // expanded rel-pos bias with -inf padding mask: [3][64][64]
  for (int i = gid; i < NH * TPAD * TPAD; i += gs) {
    int h = i / (TPAD*TPAD), r = i % (TPAD*TPAD);
    int n = r / TPAD, m = r % TPAD;
    float v;
    if (n < TOK && m < TOK) {
      int ai = n / WS_, aj = n % WS_, bi = m / WS_, bj = m % WS_;
      int idx = (ai - bi + WS_ - 1) * (2*WS_ - 1) + (aj - bj + WS_ - 1);
      v = rpb_table[idx * NH + h];
    } else {
      v = -1e30f;
    }
    rpbf[i] = v;
  }
}

// ================= main fused window-attention kernel =================
__global__ __launch_bounds__(256, 1) void wattn_main(
    const float* __restrict__ x,
    const float* __restrict__ bq, const float* __restrict__ bk, const float* __restrict__ bv,
    const float* __restrict__ conv1w, const float* __restrict__ conv2w,
    const char* __restrict__ ws,
    float* __restrict__ out, int nwin)
{
  const _Float16* wqT   = (const _Float16*)(ws + WS_WQT);
  const _Float16* wkT   = (const _Float16*)(ws + WS_WKT);
  const _Float16* wvT   = (const _Float16*)(ws + WS_WVT);
  const _Float16* wqspT = (const _Float16*)(ws + WS_WQSPT);
  const _Float16* wkspT = (const _Float16*)(ws + WS_WKSPT);
  const _Float16* wf1T  = (const _Float16*)(ws + WS_WF1T);
  const _Float16* wf2T  = (const _Float16*)(ws + WS_WF2T);
  const _Float16* wbotT = (const _Float16*)(ws + WS_WBOTT);
  const float*    biasf = (const float*)   (ws + WS_BIASF);
  const float*    rpbf  = (const float*)   (ws + WS_RPB);

  extern __shared__ __align__(32) char smem[];
  _Float16* xb  = (_Float16*)(smem + XB_OFF);
  _Float16* qb  = (_Float16*)(smem + QB_OFF);
  _Float16* kb  = (_Float16*)(smem + KB_OFF);
  _Float16* vb  = (_Float16*)(smem + VB_OFF);
  _Float16* vtb = (_Float16*)(smem + VTB_OFF);
  _Float16* qst = (_Float16*)(smem + QSTB_OFF);
  _Float16* kst = (_Float16*)(smem + KSTB_OFF);
  _Float16* spb = (_Float16*)(smem + SPB_OFF);
  _Float16* xcb = (_Float16*)(smem + XCB_OFF);
  float*    at32= (float*)   (smem + ATT32_OFF);
  _Float16* at16= (_Float16*)(smem + ATT16_OFF);
  float*    ct32= (float*)   (smem + ATT32_OFF);  // spectral scores alias (disjoint in time)
  _Float16* ct16= (_Float16*)(smem + ATT16_OFF);
  float*    c1s = (float*)   (smem + C1_OFF);
  float*    c2s = (float*)   (smem + C2_OFF);
  float*    sbq = (float*)   (smem + BQ_OFF);
  float*    sbk = (float*)   (smem + BK_OFF);
  float*    sbv = (float*)   (smem + BV_OFF);
  float*    sbf = (float*)   (smem + BF_OFF);
  _Float16* cvb = (_Float16*)(smem + ATT32_OFF); // conv mid aliases attn-f32
  _Float16* opb = qb;                            // conv out aliases q

  const int tid  = threadIdx.x;
  const int wid  = tid >> 5;
  const int lane = tid & 31;

  for (int i = tid; i < 9*CDIM; i += 256) { c1s[i] = conv1w[i]; c2s[i] = conv2w[i]; }
  for (int i = tid; i < CDIM;   i += 256) { sbq[i]=bq[i]; sbk[i]=bk[i]; sbv[i]=bv[i]; sbf[i]=biasf[i]; }
  __syncthreads();

  for (int w = blockIdx.x; w < nwin; w += gridDim.x) {
    const size_t xoff = (size_t)w * (TOK * CDIM);

    // ---- stage 1: load x -> f16, zero-pad rows 49..63 ----
    for (int i = tid; i < TPAD*CDIM; i += 256) {
      int r = i / CDIM, c = i - r*CDIM;
      xb[i] = (_Float16)((r < TOK) ? x[xoff + r*CDIM + c] : 0.0f);
    }
    // prefetch next grid-stride window's x into cache (global_prefetch_b8)
    {
      int wn = w + gridDim.x;
      if (wn < nwin) {
        const float* nx = x + (size_t)wn * (TOK * CDIM);
        for (int i = tid * 32; i < TOK*CDIM; i += 256 * 32)
          __builtin_prefetch(nx + i, 0, 3);
      }
    }
    __syncthreads();

    // ---- stage 2: q,k,v,qs,ks projections (120 WMMA tile jobs) ----
    for (int job = wid; job < 120; job += 8) {
      int which = job / 24, rem = job % 24, mt = rem / 6, nt = rem % 6;
      const _Float16* wT = (which==0)?wqT:(which==1)?wkT:(which==2)?wvT:(which==3)?wqspT:wkspT;
      v8f acc = {};
#pragma unroll
      for (int kt = 0; kt < 3; ++kt)
        acc = WMMA(a_frag(xb, mt*16, CDIM, kt*32), b_frag(wT, nt*16, CDIM, kt*32), acc);
      int n  = nt*16 + (lane & 15);
      int mb = mt*16 + ((lane >> 4) << 3);
      if (which < 3) {
        float bias = (which==0)?sbq[n]:(which==1)?sbk[n]:sbv[n];
        _Float16* dst = (which==0)?qb:(which==1)?kb:vb;
#pragma unroll
        for (int j = 0; j < 8; ++j) {
          float vv = acc[j] + bias;
          dst[(mb+j)*CDIM + n] = (_Float16)vv;
          if (which == 2) vtb[n*TPAD + (mb+j)] = (_Float16)vv;
        }
      } else {
        _Float16* dst = (which==3)?qst:kst;   // store transposed [c][tok]; pad tokens stay 0
#pragma unroll
        for (int j = 0; j < 8; ++j) dst[n*TPAD + (mb+j)] = (_Float16)acc[j];
      }
    }
    __syncthreads();

    // ---- stage 3: L2-normalize qs/ks rows over tokens ----
    for (int r = tid; r < 2*CDIM; r += 256) {
      _Float16* row = (r < CDIM) ? (qst + r*TPAD) : (kst + (r-CDIM)*TPAD);
      float s = 0.0f;
      for (int t = 0; t < TOK; ++t) { float vv = (float)row[t]; s += vv*vv; }
      float inv = 1.0f / fmaxf(sqrtf(s), 1e-12f);
      for (int t = 0; t < TOK; ++t) row[t] = (_Float16)((float)row[t] * inv);
    }
    __syncthreads();

    // ---- stage 4: spectral channel attention, all heads batched ----
    // cattn = ks @ qs^T : 3 heads x (2x2 tiles), K = 64 tokens
    for (int job = wid; job < 12; job += 8) {
      int h = job >> 2, r4 = job & 3, mt = r4 >> 1, nt = r4 & 1;
      const int hc = h * HD;
      v8f acc = {};
#pragma unroll
      for (int kt = 0; kt < 2; ++kt)
        acc = WMMA(a_frag(kst + hc*TPAD, mt*16, TPAD, kt*32),
                   b_frag(qst + hc*TPAD, nt*16, TPAD, kt*32), acc);
      int e = nt*16 + (lane&15), db = mt*16 + ((lane>>4)<<3);
#pragma unroll
      for (int j = 0; j < 8; ++j) ct32[h*HD*HD + (db+j)*HD + e] = acc[j];
    }
    __syncthreads();
    // softmax over e: 96 rows x 32, two lanes per row, logits in registers
    {
      const int r = tid >> 1;            // r = h*32 + d
      if (r < NH*HD) {
        const int part = (tid & 1) * 16;
        const float* crow = ct32 + r*HD + part;
        float vals[16];
        float mx = -1e30f;
#pragma unroll
        for (int i = 0; i < 16; ++i) { vals[i] = crow[i]*SCALE; mx = fmaxf(mx, vals[i]); }
        mx = fmaxf(mx, __shfl_xor(mx, 1, 2));
        float s = 0.0f;
#pragma unroll
        for (int i = 0; i < 16; ++i) { vals[i] = __expf(vals[i] - mx); s += vals[i]; }
        s += __shfl_xor(s, 1, 2);
        const float inv = 1.0f / s;
        _Float16* orow = ct16 + r*HD + part;
#pragma unroll
        for (int i = 0; i < 16; ++i) orow[i] = (_Float16)(vals[i] * inv);
      }
    }
    __syncthreads();
    // xc = cattn @ vs : 3 heads x (2x4 tiles), K = 32 (one k-tile)
    for (int job = wid; job < 24; job += 8) {
      int h = job >> 3, r8 = job & 7, mt = r8 >> 2, nt = r8 & 3;
      const int hc = h * HD;
      v8f acc = {};
      acc = WMMA(a_frag(ct16 + h*HD*HD, mt*16, HD, 0), b_frag(vb + hc, nt*16, CDIM, 0), acc);
      int ntok = nt*16 + (lane&15), db = mt*16 + ((lane>>4)<<3);
#pragma unroll
      for (int j = 0; j < 8; ++j) xcb[ntok*CDIM + hc + db + j] = (_Float16)acc[j];
    }
    __syncthreads();

    // ---- stage 5: spatial windowed attention, per head ----
    for (int h = 0; h < NH; ++h) {
      const int hc = h * HD;
      for (int job = wid; job < 16; job += 8) {           // scores = q @ k^T  (K=32, one k-tile)
        int mt = job >> 2, nt = job & 3;
        v8f acc = {};
        acc = WMMA(a_frag(qb + hc, mt*16, CDIM, 0), b_frag(kb + hc, nt*16, CDIM, 0), acc);
        int m = nt*16 + (lane&15), nb = mt*16 + ((lane>>4)<<3);
#pragma unroll
        for (int j = 0; j < 8; ++j) at32[(nb+j)*TPAD + m] = acc[j];
      }
      __syncthreads();
      // softmax: 64 rows x 64, four lanes per row, logits in registers (incl. rpb + pad mask)
      {
        const int r = tid >> 2, part = (tid & 3) * 16;
        const float* rp   = rpbf + (size_t)h*TPAD*TPAD + r*TPAD + part;
        const float* arow = at32 + r*TPAD + part;
        float vals[16];
        float mx = -1e30f;
#pragma unroll
        for (int i = 0; i < 16; ++i) { vals[i] = arow[i]*SCALE + rp[i]; mx = fmaxf(mx, vals[i]); }
        mx = fmaxf(mx, __shfl_xor(mx, 1, 4));
        mx = fmaxf(mx, __shfl_xor(mx, 2, 4));
        float s = 0.0f;
#pragma unroll
        for (int i = 0; i < 16; ++i) { vals[i] = __expf(vals[i] - mx); s += vals[i]; }
        s += __shfl_xor(s, 1, 4);
        s += __shfl_xor(s, 2, 4);
        const float inv = 1.0f / s;
        _Float16* orow = at16 + r*TPAD + part;
#pragma unroll
        for (int i = 0; i < 16; ++i) orow[i] = (_Float16)(vals[i] * inv);
      }
      __syncthreads();
      for (int job = wid; job < 8; job += 8) {            // sp = attn @ v  (K=64)
        int mt = job >> 1, nt = job & 1;
        v8f acc = {};
#pragma unroll
        for (int kt = 0; kt < 2; ++kt)
          acc = WMMA(a_frag(at16, mt*16, TPAD, kt*32),
                     b_frag(vtb + hc*TPAD, nt*16, TPAD, kt*32), acc);
        int d = nt*16 + (lane&15), nb = mt*16 + ((lane>>4)<<3);
#pragma unroll
        for (int j = 0; j < 8; ++j) spb[(nb+j)*CDIM + hc + d] = (_Float16)acc[j];
      }
      __syncthreads();
    }

    // ---- stage 6: depthwise conv 3x3 -> GELU -> depthwise conv 3x3 on v ----
    for (int i = tid; i < TOK*CDIM; i += 256) {
      int c = i % CDIM, t = i / CDIM, ii = t / WS_, jj = t % WS_;
      float acc = 0.0f;
#pragma unroll
      for (int di = 0; di < 3; ++di)
#pragma unroll
        for (int dj = 0; dj < 3; ++dj) {
          int yi = ii + di - 1, yj = jj + dj - 1;
          if (yi >= 0 && yi < WS_ && yj >= 0 && yj < WS_)
            acc += (float)vb[(yi*WS_+yj)*CDIM + c] * c1s[(di*3+dj)*CDIM + c];
        }
      float g = 0.5f * acc * (1.0f + erff(acc * 0.7071067811865475f));
      cvb[i] = (_Float16)g;
    }
    __syncthreads();
    for (int i = tid; i < TOK*CDIM; i += 256) {
      int c = i % CDIM, t = i / CDIM, ii = t / WS_, jj = t % WS_;
      float acc = 0.0f;
#pragma unroll
      for (int di = 0; di < 3; ++di)
#pragma unroll
        for (int dj = 0; dj < 3; ++dj) {
          int yi = ii + di - 1, yj = jj + dj - 1;
          if (yi >= 0 && yi < WS_ && yj >= 0 && yj < WS_)
            acc += (float)cvb[(yi*WS_+yj)*CDIM + c] * c2s[(di*3+dj)*CDIM + c];
        }
      opb[t*CDIM + c] = (_Float16)acc;
    }
    __syncthreads();

    // ---- stage 7: fused output projection: out = sp@Wf1 + xc@Wf2 + out_p@Wbot + biasf ----
    for (int job = wid; job < 24; job += 8) {
      int mt = job / 6, nt = job % 6;
      v8f acc = {};
#pragma unroll
      for (int kt = 0; kt < 3; ++kt) {
        acc = WMMA(a_frag(spb, mt*16, CDIM, kt*32), b_frag(wf1T,  nt*16, CDIM, kt*32), acc);
        acc = WMMA(a_frag(xcb, mt*16, CDIM, kt*32), b_frag(wf2T,  nt*16, CDIM, kt*32), acc);
        acc = WMMA(a_frag(opb, mt*16, CDIM, kt*32), b_frag(wbotT, nt*16, CDIM, kt*32), acc);
      }
      int c  = nt*16 + (lane&15);
      int nb = mt*16 + ((lane>>4)<<3);
      float bf = sbf[c];
#pragma unroll
      for (int j = 0; j < 8; ++j) {
        int row = nb + j;
        if (row < TOK) out[xoff + row*CDIM + c] = acc[j] + bf;
      }
    }
    __syncthreads();
  }
}

extern "C" void kernel_launch(void* const* d_in, const int* in_sizes, int n_in,
                              void* d_out, int out_size, void* d_ws, size_t ws_size,
                              hipStream_t stream) {
  const float* x         = (const float*)d_in[0];
  const float* rpb_table = (const float*)d_in[1];
  const float* wq        = (const float*)d_in[2];
  const float* bq        = (const float*)d_in[3];
  const float* wk        = (const float*)d_in[4];
  const float* bk        = (const float*)d_in[5];
  const float* wv        = (const float*)d_in[6];
  const float* bv        = (const float*)d_in[7];
  const float* w_ps      = (const float*)d_in[8];
  const float* b_ps      = (const float*)d_in[9];
  const float* wq_sp     = (const float*)d_in[10];
  const float* wk_sp     = (const float*)d_in[11];
  const float* w_pc      = (const float*)d_in[12];
  const float* b_pc      = (const float*)d_in[13];
  const float* conv1     = (const float*)d_in[14];
  const float* conv2     = (const float*)d_in[15];
  const float* w_proj    = (const float*)d_in[16];
  const float* b_proj    = (const float*)d_in[17];
  float* out = (float*)d_out;
  char* ws = (char*)d_ws;

  const int nwin = in_sizes[0] / (TOK * CDIM);

  wattn_prep<<<dim3(120), dim3(256), 0, stream>>>(
      rpb_table, wq, wk, wv, w_ps, b_ps, wq_sp, wk_sp, w_pc, b_pc, w_proj, b_proj, ws);

  (void)hipFuncSetAttribute((const void*)wattn_main,
                            hipFuncAttributeMaxDynamicSharedMemorySize, SMEM_BYTES);

  wattn_main<<<dim3(2048), dim3(256), SMEM_BYTES, stream>>>(
      x, bq, bk, bv, conv1, conv2, ws, out, nwin);
}
